// DGCNN_semseg_72868415144595
// MI455X (gfx1250) — matrix operand order repaired and checked by reading, not compiled, not run on hardware
//
#include <hip/hip_runtime.h>

#define B_   8
#define N_   2048
#define KNN_ 20
#define EMB_ 1024

#if defined(__has_builtin)
#if __has_builtin(__builtin_amdgcn_sched_barrier)
#define SCHED_FENCE() __builtin_amdgcn_sched_barrier(0)
#endif
#endif
#ifndef SCHED_FENCE
#define SCHED_FENCE()
#endif

typedef __attribute__((ext_vector_type(16))) unsigned short u16x16;
typedef __attribute__((ext_vector_type(8)))  unsigned short u16x8;
typedef __attribute__((ext_vector_type(16))) __bf16         v16bf;
typedef __attribute__((ext_vector_type(8)))  float          v8f;

__device__ __forceinline__ unsigned short f2bf(float f) {
  unsigned u = __builtin_bit_cast(unsigned, f);
  u += 0x7FFFu + ((u >> 16) & 1u);           // round-to-nearest-even
  return (unsigned short)(u >> 16);
}

__device__ __forceinline__ float lrelu(float v) { return v > 0.f ? v : 0.2f * v; }

__device__ __forceinline__ v8f wmma_bf16(u16x16 a, u16x16 b, v8f c) {
  return __builtin_amdgcn_wmma_f32_16x16x32_bf16(
      false, __builtin_bit_cast(v16bf, a),
      false, __builtin_bit_cast(v16bf, b),
      (short)0, c, false, false);
}

// A fragment: 16x32 (MxK) bf16, row-major [M][lda]. Two aligned 16B loads.
__device__ __forceinline__ u16x16 load_a(const unsigned short* W, int lda, int mt, int kt, int lane) {
  const unsigned short* p = W + (size_t)(mt * 16 + (lane & 15)) * lda + kt * 32 + ((lane & 16) ? 8 : 0);
  u16x8 lo = *(const u16x8*)p;
  u16x8 hi = *(const u16x8*)(p + 16);
  return __builtin_shufflevector(lo, hi, 0, 1, 2, 3, 4, 5, 6, 7, 8, 9, 10, 11, 12, 13, 14, 15);
}

// B fragment: 32x16 (KxN) bf16 stored N-major [n][ldk]. Two aligned 16B loads.
__device__ __forceinline__ u16x16 load_bT(const unsigned short* Bm, int ldk, int nt, int kt, int lane) {
  const unsigned short* p = Bm + (size_t)(nt * 16 + (lane & 15)) * ldk + kt * 32 + ((lane & 16) ? 8 : 0);
  u16x8 lo = *(const u16x8*)p;
  u16x8 hi = *(const u16x8*)(p + 16);
  return __builtin_shufflevector(lo, hi, 0, 1, 2, 3, 4, 5, 6, 7, 8, 9, 10, 11, 12, 13, 14, 15);
}

// ---------------- weight prep: fp32 -> bf16, zero padded to Mp x Kp ----------
__global__ void wprep_kernel(const float* __restrict__ W, unsigned short* __restrict__ Wbf,
                             int M, int Ks, int Mp, int Kp) {
  int i = blockIdx.x * blockDim.x + threadIdx.x;
  if (i >= Mp * Kp) return;
  int m = i / Kp, k = i % Kp;
  float v = (m < M && k < Ks) ? W[m * Ks + k] : 0.f;
  Wbf[i] = f2bf(v);
}

// ---------------- kNN prep: (B,CS,N) slice -> (B,N,C) + norms ----------------
__global__ void xtprep_kernel(const float* __restrict__ src, int CS, int co, int C,
                              float* __restrict__ xt, float* __restrict__ nrm) {
  int t = blockIdx.x * blockDim.x + threadIdx.x;
  if (t >= B_ * N_) return;
  int b = t / N_, n = t % N_;
  float s = 0.f;
  for (int c = 0; c < C; c++) {
    float v = src[(b * CS + co + c) * N_ + n];
    xt[t * C + c] = v;
    s += v * v;
  }
  nrm[t] = s;
}

// ---------------- kNN: per-thread top-20 over all columns --------------------
template <int C>
__global__ void knn_kernel(const float* __restrict__ xt, const float* __restrict__ nrm,
                           int* __restrict__ idxout) {
  int t = blockIdx.x * blockDim.x + threadIdx.x;
  if (t >= B_ * N_) return;
  float q[C];
#pragma unroll
  for (int c = 0; c < C; c++) q[c] = xt[t * C + c];
  float qn   = nrm[t];
  int   base = (t / N_) * N_;
  float tv[KNN_]; int ti[KNN_];
#pragma unroll
  for (int i = 0; i < KNN_; i++) { tv[i] = -3.4e38f; ti[i] = 0; }
  for (int j = 0; j < N_; j++) {
    const float* pj = xt + (size_t)(base + j) * C;
    float dot = 0.f;
#pragma unroll
    for (int c = 0; c < C; c++) dot += q[c] * pj[c];
    float d = 2.f * dot - qn - nrm[base + j];   // neg squared distance
    if (d > tv[KNN_ - 1]) {
      int p = KNN_ - 1;
      while (p > 0 && tv[p - 1] < d) { tv[p] = tv[p - 1]; ti[p] = ti[p - 1]; p--; }
      tv[p] = d; ti[p] = j;
    }
  }
  for (int k = 0; k < KNN_; k++) idxout[t * KNN_ + k] = ti[k];
}

// ---------------- edge-conv: gather -> WMMA conv(s) -> max over K ------------
template <int CIN, bool TWO>
struct ECfg {
  static constexpr int    KPAD = (CIN == 9) ? 32 : 128;
  static constexpr int    COLS = 80;                        // 4 points * 20 nbrs
  static constexpr size_t O_W1 = 0;
  static constexpr size_t S_W1 = (size_t)64 * KPAD * 2;
  static constexpr size_t O_W2 = O_W1 + S_W1;
  static constexpr size_t S_W2 = TWO ? (size_t)64 * 64 * 2 : 0;
  static constexpr size_t O_E  = O_W2 + S_W2;
  static constexpr size_t S_E  = (size_t)COLS * KPAD * 2;   // [col][KPAD]
  static constexpr size_t O_Y1 = O_E + S_E;
  static constexpr size_t S_Y1 = TWO ? (size_t)COLS * 64 * 2 : 0; // [col][64]
  static constexpr size_t O_YF = TWO ? O_E : (O_Y1 + S_Y1); // overlap E when 2-stage
  static constexpr size_t S_YF = (size_t)64 * COLS * 4;     // [m][col] f32
  static constexpr size_t TOT  = ((O_Y1 + S_Y1) > (O_YF + S_YF)) ? (O_Y1 + S_Y1) : (O_YF + S_YF);
};

template <int CIN, bool TWO>
__global__ void edgeconv_kernel(const float* __restrict__ X, const int* __restrict__ idx,
                                const unsigned short* __restrict__ W1bf,
                                const float* __restrict__ s1, const float* __restrict__ b1,
                                const unsigned short* __restrict__ W2bf,
                                const float* __restrict__ s2, const float* __restrict__ b2,
                                float* __restrict__ Xout) {
  using C = ECfg<CIN, TWO>;
  __shared__ __align__(16) char smem[C::TOT];
  unsigned short* usW1 = (unsigned short*)(smem + C::O_W1);
  unsigned short* usW2 = (unsigned short*)(smem + C::O_W2);
  unsigned short* usE  = (unsigned short*)(smem + C::O_E);   // [col][KPAD]
  unsigned short* usY1 = (unsigned short*)(smem + C::O_Y1);  // [col][64]
  float*          fYf  = (float*)(smem + C::O_YF);           // [m][COLS]

  int b = blockIdx.x / (N_ / 4);
  int n0 = (blockIdx.x % (N_ / 4)) * 4;
  int tid = threadIdx.x;

  // stage weights
  for (int e = tid; e < 64 * C::KPAD; e += 256) usW1[e] = W1bf[e];
  if (TWO) for (int e = tid; e < 64 * 64; e += 256) usW2[e] = W2bf[e];

  // build edge features E[col][KPAD]: k 0..CIN-1 = nb - x, CIN..2CIN-1 = x, rest 0
  for (int e = tid; e < C::KPAD * C::COLS; e += 256) {
    int c = e / C::COLS, col = e % C::COLS;
    unsigned short v = 0;
    if (c < 2 * CIN) {
      int p = col / KNN_, k = col % KNN_;
      int n = n0 + p;
      int j = idx[(b * N_ + n) * KNN_ + k];
      int ch = (c < CIN) ? c : c - CIN;
      float xc = X[((size_t)b * CIN + ch) * N_ + n];
      float val = (c < CIN) ? (X[((size_t)b * CIN + ch) * N_ + j] - xc) : xc;
      v = f2bf(val);
    }
    usE[col * C::KPAD + c] = v;
  }
  __syncthreads();

  int wv = tid >> 5, lane = tid & 31;
  // stage 1: 64 x 80 = 4 Mtiles x 5 Ntiles
  for (int job = wv; job < 20; job += 8) {
    int Mt = job / 5, Nt = job % 5;
    v8f acc = {0.f, 0.f, 0.f, 0.f, 0.f, 0.f, 0.f, 0.f};
#pragma clang loop unroll(disable)
    for (int kt = 0; kt < C::KPAD / 32; kt++) {
      acc = wmma_bf16(load_a(usW1, C::KPAD, Mt, kt, lane), load_bT(usE, C::KPAD, Nt, kt, lane), acc);
      SCHED_FENCE();
    }
#pragma unroll
    for (int r = 0; r < 8; r++) {
      int m = Mt * 16 + r + ((lane & 16) ? 8 : 0);
      int col = Nt * 16 + (lane & 15);
      float v = lrelu(acc[r] * s1[m] + b1[m]);
      if (TWO) usY1[col * 64 + m] = f2bf(v);
      else     fYf [m * C::COLS + col] = v;
    }
  }
  __syncthreads();

  if (TWO) {  // stage 2: W2 (64x64) @ Y1, output f32 into Yf (reuses E space)
    for (int job = wv; job < 20; job += 8) {
      int Mt = job / 5, Nt = job % 5;
      v8f acc = {0.f, 0.f, 0.f, 0.f, 0.f, 0.f, 0.f, 0.f};
#pragma clang loop unroll(disable)
      for (int kt = 0; kt < 2; kt++) {
        acc = wmma_bf16(load_a(usW2, 64, Mt, kt, lane), load_bT(usY1, 64, Nt, kt, lane), acc);
        SCHED_FENCE();
      }
#pragma unroll
      for (int r = 0; r < 8; r++) {
        int m = Mt * 16 + r + ((lane & 16) ? 8 : 0);
        int col = Nt * 16 + (lane & 15);
        fYf[m * C::COLS + col] = lrelu(acc[r] * s2[m] + b2[m]);
      }
    }
    __syncthreads();
  }

  // max over K neighbors -> (B,64,N)
  if (tid < 256) {
    int m = tid >> 2, p = tid & 3;
    float mx = -3.4e38f;
    for (int k = 0; k < KNN_; k++) mx = fmaxf(mx, fYf[m * C::COLS + p * KNN_ + k]);
    Xout[((size_t)b * 64 + m) * N_ + n0 + p] = mx;
  }
}

// ---------------- w4 layer (1024x192) + per-block column max -----------------
__global__ void glob_kernel(const float* __restrict__ x1, const float* __restrict__ x2,
                            const float* __restrict__ x3,
                            const unsigned short* __restrict__ w4bf,
                            const float* __restrict__ s4, const float* __restrict__ b4,
                            float* __restrict__ pmax) {
  __shared__ __align__(16) unsigned short XC[64 * 192];   // [n][192]
  __shared__ float gm[EMB_];
  int b = blockIdx.x >> 5, nt = blockIdx.x & 31;
  int n0 = nt * 64;
  int tid = threadIdx.x;

  for (int e = tid; e < 192 * 64; e += 256) {
    int c = e >> 6, nl = e & 63;       // coalesced global read, transposed LDS write
    const float* src = (c < 64) ? x1 : ((c < 128) ? x2 : x3);
    XC[nl * 192 + c] = f2bf(src[((size_t)b * 64 + (c & 63)) * N_ + n0 + nl]);
  }
  __syncthreads();

  int wv = tid >> 5, lane = tid & 31;
  // wave owns Mtiles wv*8 .. wv*8+7, processed in 2 passes of 4 to bound VGPRs
#pragma clang loop unroll(disable)
  for (int pass = 0; pass < 2; pass++) {
    int Mtb = wv * 8 + pass * 4;
    float vmax[4][8];
#pragma unroll
    for (int j = 0; j < 4; j++)
#pragma unroll
      for (int r = 0; r < 8; r++) vmax[j][r] = -3.4e38f;

#pragma clang loop unroll(disable)
    for (int Nt = 0; Nt < 4; Nt++) {
      v8f acc[4];
#pragma unroll
      for (int j = 0; j < 4; j++) acc[j] = (v8f){0.f, 0.f, 0.f, 0.f, 0.f, 0.f, 0.f, 0.f};
#pragma clang loop unroll(disable)
      for (int kt = 0; kt < 6; kt++) {
        u16x16 bf = load_bT(XC, 192, Nt, kt, lane);   // loaded once, reused 4x
#pragma unroll
        for (int j = 0; j < 4; j++)
          acc[j] = wmma_bf16(load_a(w4bf, 192, Mtb + j, kt, lane), bf, acc[j]);
        SCHED_FENCE();
      }
#pragma unroll
      for (int j = 0; j < 4; j++)
#pragma unroll
        for (int r = 0; r < 8; r++) {
          int m = (Mtb + j) * 16 + r + ((lane & 16) ? 8 : 0);
          vmax[j][r] = fmaxf(vmax[j][r], lrelu(acc[j][r] * s4[m] + b4[m]));
        }
    }
#pragma unroll
    for (int j = 0; j < 4; j++) {
#pragma unroll
      for (int r = 0; r < 8; r++) {
        for (int d = 1; d < 16; d <<= 1)
          vmax[j][r] = fmaxf(vmax[j][r], __shfl_xor(vmax[j][r], d, 32));
      }
    }
    if ((lane & 15) == 0) {
#pragma unroll
      for (int j = 0; j < 4; j++)
#pragma unroll
        for (int r = 0; r < 8; r++)
          gm[(Mtb + j) * 16 + r + ((lane & 16) ? 8 : 0)] = vmax[j][r];
    }
  }
  __syncthreads();
  for (int e = tid; e < EMB_; e += 256) pmax[((size_t)b * 32 + nt) * EMB_ + e] = gm[e];
}

__global__ void gmax_kernel(const float* __restrict__ pmax, float* __restrict__ g) {
  int t = blockIdx.x * blockDim.x + threadIdx.x;
  if (t >= B_ * EMB_) return;
  int b = t >> 10, m = t & 1023;
  float mx = -3.4e38f;
  for (int i = 0; i < 32; i++) mx = fmaxf(mx, pmax[((size_t)b * 32 + i) * EMB_ + m]);
  g[t] = mx;
}

// ---------------- fused head: w5a -> w5b -> w6 -------------------------------
__global__ void head_kernel(const float* __restrict__ g,
                            const float* __restrict__ x1, const float* __restrict__ x2,
                            const float* __restrict__ x3,
                            const unsigned short* __restrict__ w5abf,
                            const float* __restrict__ s5a, const float* __restrict__ b5a,
                            const unsigned short* __restrict__ w5bbf,
                            const float* __restrict__ s5b, const float* __restrict__ b5b,
                            const unsigned short* __restrict__ w6bf,
                            float* __restrict__ out) {
  __shared__ __align__(16) unsigned short H[16 * 1216];    // [n][1216]
  __shared__ __align__(16) unsigned short Y5A[16 * 512];   // [n][512]
  __shared__ __align__(16) unsigned short Y5B[16 * 256];   // [n][256]
  int b = blockIdx.x >> 7, nt = blockIdx.x & 127;
  int n0 = nt * 16;
  int tid = threadIdx.x;

  for (int e = tid; e < 1216 * 16; e += 256) {
    int c = e >> 4, nl = e & 15;
    float v;
    if      (c < 1024) v = g[b * EMB_ + c];
    else if (c < 1088) v = x1[((size_t)b * 64 + (c - 1024)) * N_ + n0 + nl];
    else if (c < 1152) v = x2[((size_t)b * 64 + (c - 1088)) * N_ + n0 + nl];
    else               v = x3[((size_t)b * 64 + (c - 1152)) * N_ + n0 + nl];
    H[nl * 1216 + c] = f2bf(v);
  }
  __syncthreads();

  int wv = tid >> 5, lane = tid & 31;
  // y5a: 512 rows, K=1216 (38 ktiles); kt-outer, 4 accumulators, B loaded once/kt
  {
    v8f acc[4];
#pragma unroll
    for (int j = 0; j < 4; j++) acc[j] = (v8f){0.f, 0.f, 0.f, 0.f, 0.f, 0.f, 0.f, 0.f};
#pragma clang loop unroll(disable)
    for (int kt = 0; kt < 38; kt++) {
      u16x16 bf = load_bT(H, 1216, 0, kt, lane);
#pragma unroll
      for (int j = 0; j < 4; j++)
        acc[j] = wmma_bf16(load_a(w5abf, 1216, wv * 4 + j, kt, lane), bf, acc[j]);
      SCHED_FENCE();
    }
#pragma unroll
    for (int j = 0; j < 4; j++)
#pragma unroll
      for (int r = 0; r < 8; r++) {
        int m = (wv * 4 + j) * 16 + r + ((lane & 16) ? 8 : 0);
        Y5A[(lane & 15) * 512 + m] = f2bf(lrelu(acc[j][r] * s5a[m] + b5a[m]));
      }
  }
  __syncthreads();
  // y5b: 256 rows, K=512 (16 ktiles); kt-outer, 2 accumulators
  {
    v8f acc[2];
#pragma unroll
    for (int j = 0; j < 2; j++) acc[j] = (v8f){0.f, 0.f, 0.f, 0.f, 0.f, 0.f, 0.f, 0.f};
#pragma clang loop unroll(disable)
    for (int kt = 0; kt < 16; kt++) {
      u16x16 bf = load_bT(Y5A, 512, 0, kt, lane);
#pragma unroll
      for (int j = 0; j < 2; j++)
        acc[j] = wmma_bf16(load_a(w5bbf, 512, wv * 2 + j, kt, lane), bf, acc[j]);
      SCHED_FENCE();
    }
#pragma unroll
    for (int j = 0; j < 2; j++)
#pragma unroll
      for (int r = 0; r < 8; r++) {
        int m = (wv * 2 + j) * 16 + r + ((lane & 16) ? 8 : 0);
        Y5B[(lane & 15) * 256 + m] = f2bf(lrelu(acc[j][r] * s5b[m] + b5b[m]));
      }
  }
  __syncthreads();
  // w6: 13(->16) rows, K = 256 (8 ktiles), no activation
  if (wv == 0) {
    v8f acc = {0.f, 0.f, 0.f, 0.f, 0.f, 0.f, 0.f, 0.f};
#pragma clang loop unroll(disable)
    for (int kt = 0; kt < 8; kt++) {
      acc = wmma_bf16(load_a(w6bf, 256, 0, kt, lane), load_bT(Y5B, 256, 0, kt, lane), acc);
      SCHED_FENCE();
    }
#pragma unroll
    for (int r = 0; r < 8; r++) {
      int m = r + ((lane & 16) ? 8 : 0);
      if (m < 13) out[((size_t)b * 13 + m) * N_ + n0 + (lane & 15)] = acc[r];
    }
  }
}

// ---------------------------------------------------------------------------
extern "C" void kernel_launch(void* const* d_in, const int* in_sizes, int n_in,
                              void* d_out, int out_size, void* d_ws, size_t ws_size,
                              hipStream_t stream) {
  const float* x   = (const float*)d_in[0];
  const float* w1a = (const float*)d_in[1];  const float* s1a = (const float*)d_in[2];  const float* b1a = (const float*)d_in[3];
  const float* w1b = (const float*)d_in[4];  const float* s1b = (const float*)d_in[5];  const float* b1b = (const float*)d_in[6];
  const float* w2a = (const float*)d_in[7];  const float* s2a = (const float*)d_in[8];  const float* b2a = (const float*)d_in[9];
  const float* w2b = (const float*)d_in[10]; const float* s2b = (const float*)d_in[11]; const float* b2b = (const float*)d_in[12];
  const float* w3  = (const float*)d_in[13]; const float* s3  = (const float*)d_in[14]; const float* b3  = (const float*)d_in[15];
  const float* w4  = (const float*)d_in[16]; const float* s4  = (const float*)d_in[17]; const float* b4  = (const float*)d_in[18];
  const float* w5a = (const float*)d_in[19]; const float* s5a = (const float*)d_in[20]; const float* b5a = (const float*)d_in[21];
  const float* w5b = (const float*)d_in[22]; const float* s5b = (const float*)d_in[23]; const float* b5b = (const float*)d_in[24];
  const float* w6  = (const float*)d_in[25];
  float* out = (float*)d_out;

  char* ws = (char*)d_ws;
  size_t off = 0;
  auto carve = [&](size_t bytes) { size_t r = off; off = (off + bytes + 255) & ~(size_t)255; return r; };
  unsigned short* w1a_bf = (unsigned short*)(ws + carve(64 * 32 * 2));
  unsigned short* w1b_bf = (unsigned short*)(ws + carve(64 * 64 * 2));
  unsigned short* w2a_bf = (unsigned short*)(ws + carve(64 * 128 * 2));
  unsigned short* w2b_bf = (unsigned short*)(ws + carve(64 * 64 * 2));
  unsigned short* w3_bf  = (unsigned short*)(ws + carve(64 * 128 * 2));
  unsigned short* w4_bf  = (unsigned short*)(ws + carve((size_t)EMB_ * 192 * 2));
  unsigned short* w5a_bf = (unsigned short*)(ws + carve((size_t)512 * 1216 * 2));
  unsigned short* w5b_bf = (unsigned short*)(ws + carve((size_t)256 * 512 * 2));
  unsigned short* w6_bf  = (unsigned short*)(ws + carve(16 * 256 * 2));
  float* xt   = (float*)(ws + carve((size_t)B_ * N_ * 64 * 4));
  float* nrm  = (float*)(ws + carve((size_t)B_ * N_ * 4));
  int*   idxw = (int*)  (ws + carve((size_t)B_ * N_ * KNN_ * 4));
  float* x1b  = (float*)(ws + carve((size_t)B_ * 64 * N_ * 4));
  float* x2b  = (float*)(ws + carve((size_t)B_ * 64 * N_ * 4));
  float* x3b  = (float*)(ws + carve((size_t)B_ * 64 * N_ * 4));
  float* pmax = (float*)(ws + carve((size_t)B_ * 32 * EMB_ * 4));
  float* gbuf = (float*)(ws + carve((size_t)B_ * EMB_ * 4));

  auto WPREP = [&](const float* src, unsigned short* dst, int M, int Ks, int Mp, int Kp) {
    int tot = Mp * Kp;
    wprep_kernel<<<(tot + 255) / 256, 256, 0, stream>>>(src, dst, M, Ks, Mp, Kp);
  };
  WPREP(w1a, w1a_bf, 64, 18, 64, 32);
  WPREP(w1b, w1b_bf, 64, 64, 64, 64);
  WPREP(w2a, w2a_bf, 64, 128, 64, 128);
  WPREP(w2b, w2b_bf, 64, 64, 64, 64);
  WPREP(w3,  w3_bf,  64, 128, 64, 128);
  WPREP(w4,  w4_bf,  EMB_, 192, EMB_, 192);
  WPREP(w5a, w5a_bf, 512, 1216, 512, 1216);
  WPREP(w5b, w5b_bf, 256, 512, 256, 512);
  WPREP(w6,  w6_bf,  13, 256, 16, 256);

  const int PT = (B_ * N_) / 256;        // 64 blocks
  const int EC = B_ * (N_ / 4);          // 4096 blocks

  // layer 1: knn on x[:,6:9], edge conv on all 9 channels
  xtprep_kernel<<<PT, 256, 0, stream>>>(x, 9, 6, 3, xt, nrm);
  knn_kernel<3><<<PT, 256, 0, stream>>>(xt, nrm, idxw);
  edgeconv_kernel<9, true><<<EC, 256, 0, stream>>>(x, idxw, w1a_bf, s1a, b1a, w1b_bf, s1b, b1b, x1b);

  // layer 2
  xtprep_kernel<<<PT, 256, 0, stream>>>(x1b, 64, 0, 64, xt, nrm);
  knn_kernel<64><<<PT, 256, 0, stream>>>(xt, nrm, idxw);
  edgeconv_kernel<64, true><<<EC, 256, 0, stream>>>(x1b, idxw, w2a_bf, s2a, b2a, w2b_bf, s2b, b2b, x2b);

  // layer 3 (single conv)
  xtprep_kernel<<<PT, 256, 0, stream>>>(x2b, 64, 0, 64, xt, nrm);
  knn_kernel<64><<<PT, 256, 0, stream>>>(xt, nrm, idxw);
  edgeconv_kernel<64, false><<<EC, 256, 0, stream>>>(x2b, idxw, w3_bf, s3, b3, nullptr, nullptr, nullptr, x3b);

  // global feature
  glob_kernel<<<B_ * 32, 256, 0, stream>>>(x1b, x2b, x3b, w4_bf, s4, b4, pmax);
  gmax_kernel<<<(B_ * EMB_) / 256, 256, 0, stream>>>(pmax, gbuf);

  // head
  head_kernel<<<B_ * (N_ / 16), 256, 0, stream>>>(gbuf, x1b, x2b, x3b,
                                                  w5a_bf, s5a, b5a,
                                                  w5b_bf, s5b, b5b,
                                                  w6_bf, out);
}
